// MultiHeadAttendtion_KVCache_83202106458252
// MI455X (gfx1250) — matrix operand order repaired and checked by reading, not compiled, not run on hardware
//
#include <hip/hip_runtime.h>
#include <hip/hip_bf16.h>

// ---------------- problem constants ----------------
static constexpr int Bn = 2;
static constexpr int Sn = 2048;
static constexpr int Dn = 1024;
static constexpr int Hn = 16;
static constexpr int HDn = 64;           // Dn / Hn
static constexpr int BS = Bn * Sn;       // 4096 rows

// ---------------- vector types ----------------
typedef __bf16 bf16;
typedef __attribute__((ext_vector_type(4)))  bf16  v4bf;
typedef __attribute__((ext_vector_type(8)))  bf16  v8bf;
typedef __attribute__((ext_vector_type(16))) bf16  v16bf;
typedef __attribute__((ext_vector_type(8)))  float v8f;

static __device__ __forceinline__ v8f wmma_bf16(v16bf a, v16bf b, v8f c) {
  // (neg_a, A, neg_b, B, c_mod, C, reuse_a, reuse_b)
  return __builtin_amdgcn_wmma_f32_16x16x32_bf16(false, a, false, b, (short)0, c, false, false);
}

static __device__ __forceinline__ v8bf ld8(const bf16* p) {
  return *(const v8bf*)p;                    // 16-byte load (ds/global b128)
}
static __device__ __forceinline__ v16bf cat8(v8bf lo, v8bf hi) {
  return __builtin_shufflevector(lo, hi, 0,1,2,3,4,5,6,7,8,9,10,11,12,13,14,15);
}

// ============================================================================
// Kernel 1 (templated on WHICH): out = x @ W^T
//   WHICH=0: Q -> Qh[b,h,s,hd]  (bf16)
//   WHICH=1: K -> Kh[b,h,s,hd]  (bf16, * mask[b,s])
//   WHICH=2: V -> Vt[b,h,hd,s]  (bf16 transposed, * mask[b,s])
// ============================================================================
template <int WHICH>
__global__ __launch_bounds__(256)
void qkv_proj_kernel(const float* __restrict__ x, const float* __restrict__ mask,
                     const float* __restrict__ W, bf16* __restrict__ dst)
{
  __shared__ __attribute__((aligned(32))) bf16 As[128][40];   // padded rows (20 DW stride)
  __shared__ __attribute__((aligned(32))) bf16 Bs[64][40];
  __shared__ __attribute__((aligned(32))) bf16 Es[8][1536];   // epilogue staging

  const int tid  = threadIdx.x;
  const int lane = tid & 31;
  const int wave = tid >> 5;
  const int half = lane >> 4;
  const int r0   = blockIdx.x * 128;     // output rows (over B*S)
  const int n0   = blockIdx.y * 64;      // output cols (over D) == one head

  v8f acc[4] = {};

  for (int kk = 0; kk < Dn; kk += 32) {
    __syncthreads();
    // ---- stage A tile (128x32) f32 -> bf16 ----
#pragma unroll
    for (int p = 0; p < 4; ++p) {
      const int row = p * 32 + (tid >> 3);
      const int kc  = (tid & 7) << 2;
      const float4 f = *(const float4*)(x + (size_t)(r0 + row) * Dn + kk + kc);
      v4bf t = { (bf16)f.x, (bf16)f.y, (bf16)f.z, (bf16)f.w };
      *(v4bf*)&As[row][kc] = t;
    }
    // ---- stage B tile (64x32 rows of W) f32 -> bf16 ----
    {
      const int row = tid >> 2;
      const int kc  = (tid & 3) << 3;
      const float* src = W + (size_t)(n0 + row) * Dn + kk + kc;
      const float4 f0 = *(const float4*)src;
      const float4 f1 = *(const float4*)(src + 4);
      v8bf t = { (bf16)f0.x, (bf16)f0.y, (bf16)f0.z, (bf16)f0.w,
                 (bf16)f1.x, (bf16)f1.y, (bf16)f1.z, (bf16)f1.w };
      *(v8bf*)&Bs[row][kc] = t;
    }
    __syncthreads();

    // ---- WMMA: wave owns rows [wave*16, wave*16+16), cols n0..n0+63 ----
    const int mrow = wave * 16 + (lane & 15);
    const int ks   = half << 4;                       // 0 or 16
    const v16bf a = cat8(ld8(&As[mrow][ks]), ld8(&As[mrow][ks + 8]));
    v16bf bf[4];
#pragma unroll
    for (int t = 0; t < 4; ++t)
      bf[t] = *(const v16bf*)&Bs[t * 16 + (lane & 15)][half << 4];
#pragma unroll
    for (int t = 0; t < 4; ++t)
      acc[t] = wmma_bf16(a, bf[t], acc[t]);
  }

  // ---- epilogue: C-layout -> LDS -> coalesced 32B row stores ----
  const int bidx  = r0 / Sn;                 // whole tile inside one batch
  const int srow0 = (r0 & (Sn - 1)) + wave * 16;
  const int h     = n0 >> 6;

  float mval[8];
#pragma unroll
  for (int i = 0; i < 8; ++i)
    mval[i] = (WHICH == 0) ? 1.0f : mask[bidx * Sn + srow0 + i + (half << 3)];

#pragma unroll
  for (int t = 0; t < 4; ++t)
#pragma unroll
    for (int i = 0; i < 8; ++i) {
      const int m = i + (half << 3);
      const int d = t * 16 + (lane & 15);
      const bf16 v = (bf16)(acc[t][i] * mval[i]);
      if (WHICH == 2) Es[wave][d * 24 + m] = v;   // [hd][s] rows (stride 24)
      else            Es[wave][m * 72 + d] = v;   // [s][hd] rows (stride 72)
    }
  // same-wave LDS: in-order, no barrier needed
  if (WHICH == 2) {
#pragma unroll
    for (int c = 0; c < 2; ++c) {
      const int d = lane + (c << 5);
      bf16* p = dst + ((size_t)(bidx * Hn + h) * HDn + d) * Sn + srow0;
      *(v16bf*)p = *(const v16bf*)&Es[wave][d * 24];
    }
  } else {
    const int m = lane & 15;
    bf16* p = dst + (((size_t)(bidx * Hn + h) * Sn + srow0 + m) << 6);
#pragma unroll
    for (int c = 0; c < 2; ++c) {
      const int off = ((half << 1) + c) << 4;     // 4 chunks of 16 bf16
      *(v16bf*)(p + off) = *(const v16bf*)&Es[wave][m * 72 + off];
    }
  }
}

// ============================================================================
// Kernel 2: causal flash attention, block-cooperative K/V staging.
//   Block: 128 q rows (16 per wave).  K/V tiles of 64 keys staged in LDS.
//   Ctxb[b,s,h*64+d] (bf16) = softmax(QK^T/8) @ V
// ============================================================================
__global__ __launch_bounds__(256)
void attn_kernel(const bf16* __restrict__ Qh, const bf16* __restrict__ Kh,
                 const bf16* __restrict__ Vt, bf16* __restrict__ Ctxb)
{
  __shared__ __attribute__((aligned(32))) bf16 Kt [64][72];     // keys x hd (padded)
  __shared__ __attribute__((aligned(32))) bf16 Vtl[64][72];     // hd x keys (padded)
  __shared__ __attribute__((aligned(32))) bf16 Ps[8][16][72];   // wave-private P tiles

  const int tid  = threadIdx.x;
  const int lane = tid & 31;
  const int wave = tid >> 5;
  const int half = lane >> 4;
  const int q0   = blockIdx.x * 128 + wave * 16;
  const int qmax = blockIdx.x * 128 + 127;       // block-wide causal bound
  const int bh   = blockIdx.y;
  const int b = bh >> 4, h = bh & 15;

  const bf16* __restrict__ Qb = Qh + (size_t)bh * Sn * 64;
  const bf16* __restrict__ Kb = Kh + (size_t)bh * Sn * 64;
  const bf16* __restrict__ Vb = Vt + (size_t)bh * 64 * Sn;

  const int ks = half << 4;                  // A-frag K chunk select (0 or 16)

  // persistent Q A-fragments (16x64 split into two K=32 frags)
  const bf16* qp = Qb + (size_t)(q0 + (lane & 15)) * 64;
  const v16bf aq0 = cat8(ld8(qp + ks),      ld8(qp + ks + 8));
  const v16bf aq1 = cat8(ld8(qp + 32 + ks), ld8(qp + 32 + ks + 8));

  v8f ctx[4] = {};
  float mrow[8], lrow[8];
#pragma unroll
  for (int i = 0; i < 8; ++i) { mrow[i] = -3.0e38f; lrow[i] = 0.0f; }

  for (int k0 = 0; k0 <= qmax; k0 += 64) {
    __syncthreads();                           // previous tile fully consumed
    // ---- cooperative stage: 64 keys of K and V^T (coalesced 32B loads) ----
    {
      const int row = tid >> 2;                // 0..63
      const int kc  = (tid & 3) << 4;          // 0,16,32,48
      *(v16bf*)&Kt [row][kc] = *(const v16bf*)(Kb + (size_t)(k0 + row) * 64 + kc);
      *(v16bf*)&Vtl[row][kc] = *(const v16bf*)(Vb + (size_t)row * Sn + k0 + kc);
    }
    __syncthreads();
    if (k0 > q0 + 15) continue;                // wave-uniform: past causal range

    // ---- scores: batch 8 K-frags from LDS, then 8 WMMA ----
    v16bf kf[4][2];
#pragma unroll
    for (int t = 0; t < 4; ++t) {
      const bf16* kp = &Kt[t * 16 + (lane & 15)][half << 4];
      kf[t][0] = *(const v16bf*)kp;
      kf[t][1] = *(const v16bf*)(kp + 32);
    }
    v8f sc[4];
#pragma unroll
    for (int t = 0; t < 4; ++t) {
      v8f z = {};
      z = wmma_bf16(aq0, kf[t][0], z);
      sc[t] = wmma_bf16(aq1, kf[t][1], z);
    }
    // ---- scale 1/sqrt(64) + causal mask ----
#pragma unroll
    for (int t = 0; t < 4; ++t)
#pragma unroll
      for (int i = 0; i < 8; ++i) {
        const int kcol = k0 + t * 16 + (lane & 15);
        const int qr   = q0 + i + (half << 3);
        const float sv = sc[t][i] * 0.125f;
        sc[t][i] = (kcol > qr) ? -3.0e38f : sv;
      }
    // ---- online softmax: row max across 16 lanes of the half-wave ----
    float mx[8];
#pragma unroll
    for (int i = 0; i < 8; ++i)
      mx[i] = fmaxf(fmaxf(sc[0][i], sc[1][i]), fmaxf(sc[2][i], sc[3][i]));
#pragma unroll
    for (int off = 8; off >= 1; off >>= 1)
#pragma unroll
      for (int i = 0; i < 8; ++i)
        mx[i] = fmaxf(mx[i], __shfl_xor(mx[i], off, 32));

    float corr[8], rs[8];
#pragma unroll
    for (int i = 0; i < 8; ++i) {
      const float mn = fmaxf(mrow[i], mx[i]);
      corr[i] = __expf(mrow[i] - mn);
      mrow[i] = mn;
      rs[i]   = 0.0f;
    }
    // ---- p = exp(s - m), row sums, stash P (C-layout -> LDS) ----
#pragma unroll
    for (int t = 0; t < 4; ++t)
#pragma unroll
      for (int i = 0; i < 8; ++i) {
        const float p = __expf(sc[t][i] - mrow[i]);
        rs[i] += p;
        Ps[wave][i + (half << 3)][t * 16 + (lane & 15)] = (bf16)p;
      }
#pragma unroll
    for (int off = 8; off >= 1; off >>= 1)
#pragma unroll
      for (int i = 0; i < 8; ++i)
        rs[i] += __shfl_xor(rs[i], off, 32);
#pragma unroll
    for (int i = 0; i < 8; ++i)
      lrow[i] = lrow[i] * corr[i] + rs[i];
    // ---- rescale ctx accumulators ----
#pragma unroll
    for (int t = 0; t < 4; ++t)
#pragma unroll
      for (int i = 0; i < 8; ++i)
        ctx[t][i] *= corr[i];

    // ---- reload P as A-fragments (transpose via LDS, same-wave in-order DS) ----
    const bf16* prow = &Ps[wave][lane & 15][0];
    const v16bf pa0 = cat8(ld8(prow + ks),      ld8(prow + ks + 8));
    const v16bf pa1 = cat8(ld8(prow + 32 + ks), ld8(prow + 32 + ks + 8));

    // ---- ctx += P @ V: batch 8 V-frags from LDS, then 8 WMMA ----
    v16bf vf[4][2];
#pragma unroll
    for (int t = 0; t < 4; ++t) {
      const bf16* vp = &Vtl[t * 16 + (lane & 15)][half << 4];
      vf[t][0] = *(const v16bf*)vp;
      vf[t][1] = *(const v16bf*)(vp + 32);
    }
#pragma unroll
    for (int t = 0; t < 4; ++t) {
      ctx[t] = wmma_bf16(pa0, vf[t][0], ctx[t]);
      ctx[t] = wmma_bf16(pa1, vf[t][1], ctx[t]);
    }
  }

  // ---- normalize, stage through Ps, write coalesced bf16 rows ----
#pragma unroll
  for (int t = 0; t < 4; ++t)
#pragma unroll
    for (int i = 0; i < 8; ++i)
      Ps[wave][i + (half << 3)][t * 16 + (lane & 15)] = (bf16)(ctx[t][i] / lrow[i]);

  const int m = lane & 15;
  bf16* p = Ctxb + (size_t)(b * Sn + q0 + m) * Dn + h * 64;
#pragma unroll
  for (int c = 0; c < 2; ++c) {
    const int off = ((half << 1) + c) << 4;
    *(v16bf*)(p + off) = *(const v16bf*)&Ps[wave][m][off];
  }
}

// ============================================================================
// Kernel 3: output projection  out = Ctxb @ Wo^T + b_o   (f32 out)
// ============================================================================
__global__ __launch_bounds__(256)
void out_proj_kernel(const bf16* __restrict__ Ctxb, const float* __restrict__ Wo,
                     const float* __restrict__ bo, float* __restrict__ out)
{
  __shared__ __attribute__((aligned(32))) bf16  As[128][40];
  __shared__ __attribute__((aligned(32))) bf16  Bs[64][40];
  __shared__ __attribute__((aligned(32))) float Ef[8][1088];   // 16 rows x 68 f32

  const int tid  = threadIdx.x;
  const int lane = tid & 31;
  const int wave = tid >> 5;
  const int half = lane >> 4;
  const int r0   = blockIdx.x * 128;
  const int n0   = blockIdx.y * 64;

  v8f acc[4] = {};

  for (int kk = 0; kk < Dn; kk += 32) {
    __syncthreads();
    // ---- A tile: straight 32B bf16 copy ----
    {
      const int row = tid >> 1;
      const int kc  = (tid & 1) << 4;
      *(v16bf*)&As[row][kc] = *(const v16bf*)(Ctxb + (size_t)(r0 + row) * Dn + kk + kc);
    }
    // ---- B tile: f32 -> bf16 ----
    {
      const int row = tid >> 2;
      const int kc  = (tid & 3) << 3;
      const float* src = Wo + (size_t)(n0 + row) * Dn + kk + kc;
      const float4 f0 = *(const float4*)src;
      const float4 f1 = *(const float4*)(src + 4);
      v8bf t = { (bf16)f0.x, (bf16)f0.y, (bf16)f0.z, (bf16)f0.w,
                 (bf16)f1.x, (bf16)f1.y, (bf16)f1.z, (bf16)f1.w };
      *(v8bf*)&Bs[row][kc] = t;
    }
    __syncthreads();

    const int mrow = wave * 16 + (lane & 15);
    const int ksl  = half << 4;
    const v16bf a = cat8(ld8(&As[mrow][ksl]), ld8(&As[mrow][ksl + 8]));
    v16bf bf[4];
#pragma unroll
    for (int t = 0; t < 4; ++t)
      bf[t] = *(const v16bf*)&Bs[t * 16 + (lane & 15)][half << 4];
#pragma unroll
    for (int t = 0; t < 4; ++t)
      acc[t] = wmma_bf16(a, bf[t], acc[t]);
  }

  // ---- epilogue: bias, stage f32 tile in LDS, coalesced float4 row stores ----
#pragma unroll
  for (int t = 0; t < 4; ++t) {
    const float bias = bo[n0 + t * 16 + (lane & 15)];
#pragma unroll
    for (int i = 0; i < 8; ++i)
      Ef[wave][(i + (half << 3)) * 68 + t * 16 + (lane & 15)] = acc[t][i] + bias;
  }
  const int m = lane & 15;
  float* orow = out + (size_t)(r0 + wave * 16 + m) * Dn + n0;
  const float* srow = &Ef[wave][m * 68];
#pragma unroll
  for (int j = 0; j < 8; ++j) {
    const int sl = (half << 3) + j;          // 16 float4 slots per row
    *(float4*)(orow + (sl << 2)) = *(const float4*)(srow + (sl << 2));
  }
}

// ============================================================================
// launcher
// ============================================================================
extern "C" void kernel_launch(void* const* d_in, const int* in_sizes, int n_in,
                              void* d_out, int out_size, void* d_ws, size_t ws_size,
                              hipStream_t stream) {
  (void)in_sizes; (void)n_in; (void)out_size; (void)ws_size;

  const float* x    = (const float*)d_in[0];
  const float* mask = (const float*)d_in[1];
  const float* Wq   = (const float*)d_in[2];
  const float* Wk   = (const float*)d_in[3];
  const float* Wv   = (const float*)d_in[4];
  const float* Wo   = (const float*)d_in[5];
  const float* bo   = (const float*)d_in[6];
  float* out        = (float*)d_out;

  // workspace layout (32 MB): Qh 8MB | Kh 8MB | Vt 8MB | Ctxb 8MB
  char* ws = (char*)d_ws;
  bf16* Qh   = (bf16*)(ws);
  bf16* Kh   = (bf16*)(ws + (size_t)(8u  << 20));
  bf16* Vt   = (bf16*)(ws + (size_t)(16u << 20));
  bf16* Ctxb = (bf16*)(ws + (size_t)(24u << 20));

  dim3 g1(BS / 128, Dn / 64);
  qkv_proj_kernel<0><<<g1, 256, 0, stream>>>(x, mask, Wq, Qh);
  qkv_proj_kernel<1><<<g1, 256, 0, stream>>>(x, mask, Wk, Kh);
  qkv_proj_kernel<2><<<g1, 256, 0, stream>>>(x, mask, Wv, Vt);

  dim3 g2(Sn / 128, Bn * Hn);
  attn_kernel<<<g2, 256, 0, stream>>>(Qh, Kh, Vt, Ctxb);

  dim3 g3(BS / 128, Dn / 64);
  out_proj_kernel<<<g3, 256, 0, stream>>>(Ctxb, Wo, bo, out);
}
